// GraphNet_31456340476600
// MI455X (gfx1250) — matrix operand order repaired and checked by reading, not compiled, not run on hardware
//
#include <hip/hip_runtime.h>
#include <hip/hip_bf16.h>

#define NN 131072
#define EE 131072
#define RR 8
#define GG 4096
#define HID 128

typedef __attribute__((ext_vector_type(16))) __bf16 v16bf;
typedef __attribute__((ext_vector_type(8)))  __bf16 v8bf;
typedef __attribute__((ext_vector_type(8)))  float  v8f;

__device__ __forceinline__ __bf16 f2bf(float f) {
    union { float f; unsigned u; } in; in.f = f;
    unsigned u = in.u;
    unsigned r = u + 0x7fffu + ((u >> 16) & 1u);   // round-to-nearest-even
    union { unsigned short s; __bf16 b; } out; out.s = (unsigned short)(r >> 16);
    return out.b;
}

// ---------------------------------------------------------------------------
// Weight prep: f32 (K x ncols) row-major -> bf16 (ncols x 64) "transposed"
// so each WMMA B-fragment lane reads 16 contiguous K values of one column.
// ---------------------------------------------------------------------------
__global__ void transpose_w_bf16(const float* __restrict__ W, __bf16* __restrict__ WT, int ncols) {
    int i = blockIdx.x * blockDim.x + threadIdx.x;
    if (i >= ncols * 64) return;
    int n = i >> 6, k = i & 63;
    WT[i] = f2bf(W[k * ncols + n]);
}

// Block-diagonal 64x64 (transposed, bf16) from a_rel/m_rel[L][R][H][D][D].
__global__ void build_bd_bf16(const float* __restrict__ a_rel, const float* __restrict__ m_rel,
                              __bf16* __restrict__ bd) {
    int i = blockIdx.x * blockDim.x + threadIdx.x;
    if (i >= 2 * RR * 2 * 4096) return;       // L*R*2 mats of 64x64
    int t = i >> 12;                          // t = (l*R+r)*2 + which
    int idx = i & 4095;
    int which = t & 1, lr = t >> 1;
    int nout = idx >> 6, kk = idx & 63;       // BDT[nout][kk]
    int ho = nout >> 4, e = nout & 15, hi = kk >> 4, d = kk & 15;
    float v = 0.f;
    if (ho == hi) {
        const float* rel = (which == 0) ? a_rel : m_rel;
        v = rel[((lr * 4 + ho) * 16 + d) * 16 + e];
    }
    bd[i] = f2bf(v);
}

// ---------------------------------------------------------------------------
// Input projection: h = x @ in_W(3x64) + b ; writes f32 + bf16
// ---------------------------------------------------------------------------
__global__ void input_proj(const float* __restrict__ x, const float* __restrict__ W,
                           const float* __restrict__ b, float* __restrict__ h,
                           __bf16* __restrict__ hbf) {
    int n = blockIdx.x * blockDim.x + threadIdx.x;
    if (n >= NN) return;
    float x0 = x[n * 3], x1 = x[n * 3 + 1], x2 = x[n * 3 + 2];
    #pragma unroll 4
    for (int c = 0; c < 64; ++c) {
        float v = fmaf(x0, W[c], fmaf(x1, W[64 + c], fmaf(x2, W[128 + c], b[c])));
        h[(size_t)n * 64 + c] = v;
        hbf[(size_t)n * 64 + c] = f2bf(v);
    }
}

// ---------------------------------------------------------------------------
// WMMA GEMM: Y(M x NTILES*16) = A_bf16(M x 64) @ W + bias, 16-row tile/wave.
// All B fragments pre-loaded (one clause) so the WMMA chain drains with
// staggered loadcnt waits. Epilogue fully compile-time specialized.
// FLAGS: 1 = relu, 2 = sigmoid(skip) blend with hold (NTILES must be 4)
// ---------------------------------------------------------------------------
template <int NTILES, int FLAGS, bool HAS_BIAS, bool HAS_F, bool HAS_B>
__global__ void gemm64_wmma(const __bf16* __restrict__ A, const __bf16* __restrict__ WT,
                            const float* __restrict__ bias,
                            float* __restrict__ outF, __bf16* __restrict__ outB,
                            const float* __restrict__ hold, const float* __restrict__ skip_g) {
    constexpr int ncols = NTILES * 16;
    int m0 = blockIdx.x * 16;
    int lane = threadIdx.x;            // blockDim == 32
    int hs = lane >> 4;                // half select
    int lm = lane & 15;
    int row = m0 + lm;

    // A fragments (16-bit A 16x32): lanes<16 K{0..15}, lanes>=16 K{16..31}
    v16bf afrag[2];
    #pragma unroll
    for (int kt = 0; kt < 2; ++kt) {
        const __bf16* ap = A + (size_t)row * 64 + kt * 32 + hs * 16;
        v8bf c0 = *(const v8bf*)(ap);
        v8bf c1 = *(const v8bf*)(ap + 8);
        v16bf f;
        #pragma unroll
        for (int j = 0; j < 8; ++j) { f[j] = c0[j]; f[8 + j] = c1[j]; }
        afrag[kt] = f;
    }

    // B fragments: lane<16 holds col, K kt*32..+15 ; lane>=16 K kt*32+16..+31
    v16bf bfrag[NTILES][2];
    #pragma unroll
    for (int nt = 0; nt < NTILES; ++nt) {
        int col = nt * 16 + lm;
        #pragma unroll
        for (int kt = 0; kt < 2; ++kt)
            bfrag[nt][kt] = *(const v16bf*)(WT + (size_t)col * 64 + kt * 32 + hs * 16);
    }

    v8f acc[NTILES];
    #pragma unroll
    for (int nt = 0; nt < NTILES; ++nt) {
        v8f c = {};
        c = __builtin_amdgcn_wmma_f32_16x16x32_bf16(false, afrag[0], false, bfrag[nt][0],
                                                    (short)0, c, false, false);
        c = __builtin_amdgcn_wmma_f32_16x16x32_bf16(false, afrag[1], false, bfrag[nt][1],
                                                    (short)0, c, false, false);
        acc[nt] = c;
    }

    float g = 0.f, og = 0.f;
    if constexpr ((FLAGS & 2) != 0) { g = 1.f / (1.f + __expf(-*skip_g)); og = 1.f - g; }

    #pragma unroll
    for (int nt = 0; nt < NTILES; ++nt) {
        int col = nt * 16 + lm;
        float bv = 0.f;
        if constexpr (HAS_BIAS) bv = bias[col];
        #pragma unroll
        for (int j = 0; j < 8; ++j) {
            int rrow = m0 + j + hs * 8;            // C/D: VGPR j -> M = j (+8 hi lanes)
            float val = acc[nt][j] + bv;
            if constexpr ((FLAGS & 1) != 0) val = fmaxf(val, 0.f);
            if constexpr ((FLAGS & 2) != 0) val = g * val + og * hold[(size_t)rrow * ncols + col];
            if constexpr (HAS_F) outF[(size_t)rrow * ncols + col] = val;
            if constexpr (HAS_B) outB[(size_t)rrow * ncols + col] = f2bf(val);
        }
    }
}

// ---------------------------------------------------------------------------
// Edge pass 1: logits + encoded segment-max (one relation). float4 gathers.
// ---------------------------------------------------------------------------
__global__ void edge_logits(const int* __restrict__ eix, const float* __restrict__ q,
                            const float* __restrict__ krel, const float* __restrict__ prel,
                            float* __restrict__ logits_r, unsigned* __restrict__ maxbuf) {
    int e = blockIdx.x * blockDim.x + threadIdx.x;
    if (e >= EE) return;
    int src = eix[e], dst = eix[EE + e];
    const float4* qp = (const float4*)(q + (size_t)dst * 64);
    const float4* kp = (const float4*)(krel + (size_t)src * 64);
    float lg4[4];
    #pragma unroll
    for (int h = 0; h < 4; ++h) {
        float acc = 0.f;
        #pragma unroll
        for (int d4 = 0; d4 < 4; ++d4) {
            float4 qv = qp[h * 4 + d4], kv = kp[h * 4 + d4];
            acc = fmaf(qv.x, kv.x, acc); acc = fmaf(qv.y, kv.y, acc);
            acc = fmaf(qv.z, kv.z, acc); acc = fmaf(qv.w, kv.w, acc);
        }
        float lg = acc * prel[h] * 0.25f;                 // inv_sqrt(D=16)
        lg4[h] = lg;
        unsigned enc = __float_as_uint(lg);
        enc = (enc & 0x80000000u) ? ~enc : (enc | 0x80000000u);
        atomicMax(&maxbuf[dst * 4 + h], enc);
    }
    *(float4*)(logits_r + (size_t)e * 4) = *(const float4*)lg4;
}

// Edge pass 2 (all relations): ex = exp(l - m), accumulate denominators
__global__ void edge_softmax_den(const int* __restrict__ eidx_all, const unsigned* __restrict__ maxbuf,
                                 float* __restrict__ logits, float* __restrict__ den) {
    int i = blockIdx.x * blockDim.x + threadIdx.x;
    if (i >= RR * EE) return;
    int r = i >> 17, e = i & (EE - 1);
    int dst = eidx_all[(r * 2 + 1) * EE + e];
    float l4[4];
    *(float4*)l4 = *(const float4*)(logits + (size_t)i * 4);
    #pragma unroll
    for (int h = 0; h < 4; ++h) {
        unsigned enc = maxbuf[dst * 4 + h];
        unsigned dec = (enc & 0x80000000u) ? (enc & 0x7fffffffu) : ~enc;
        float ex = __expf(l4[h] - __uint_as_float(dec));
        l4[h] = ex;
        atomicAdd(&den[dst * 4 + h], ex);
    }
    *(float4*)(logits + (size_t)i * 4) = *(const float4*)l4;
}

// Edge pass 3 (one relation): scatter attn-weighted v_rel into agg
__global__ void edge_agg(const int* __restrict__ eix, const float* __restrict__ ex_r,
                         const float* __restrict__ den, const float* __restrict__ vrel,
                         float* __restrict__ agg) {
    int e = blockIdx.x * blockDim.x + threadIdx.x;
    if (e >= EE) return;
    int src = eix[e], dst = eix[EE + e];
    const float4* vp = (const float4*)(vrel + (size_t)src * 64);
    float* ap = agg + (size_t)dst * 64;
    float a4[4];
    *(float4*)a4 = *(const float4*)(ex_r + (size_t)e * 4);
    #pragma unroll
    for (int h = 0; h < 4; ++h) {
        float attn = a4[h] / fmaxf(den[dst * 4 + h], 1e-16f);
        #pragma unroll
        for (int d4 = 0; d4 < 4; ++d4) {
            float4 vv = vp[h * 4 + d4];
            int base = h * 16 + d4 * 4;
            atomicAdd(&ap[base + 0], vv.x * attn);
            atomicAdd(&ap[base + 1], vv.y * attn);
            atomicAdd(&ap[base + 2], vv.z * attn);
            atomicAdd(&ap[base + 3], vv.w * attn);
        }
    }
}

// gelu(tanh approx) -> bf16
__global__ void gelu_to_bf16(const float* __restrict__ in, __bf16* __restrict__ out, int n) {
    int i = blockIdx.x * blockDim.x + threadIdx.x;
    if (i >= n) return;
    float v = in[i];
    float u = 0.7978845608028654f * (v + 0.044715f * v * v * v);
    out[i] = f2bf(0.5f * v * (1.f + tanhf(u)));
}

// Segment mean pool
__global__ void pool_accum(const float* __restrict__ h, const int* __restrict__ batch,
                           float* __restrict__ sums, float* __restrict__ cnt) {
    int n = blockIdx.x * blockDim.x + threadIdx.x;
    if (n >= NN) return;
    int b = batch[n];
    atomicAdd(&cnt[b], 1.0f);
    const float4* hp = (const float4*)(h + (size_t)n * 64);
    float* sp = sums + (size_t)b * 64;
    #pragma unroll
    for (int c4 = 0; c4 < 16; ++c4) {
        float4 hv = hp[c4];
        atomicAdd(&sp[c4 * 4 + 0], hv.x);
        atomicAdd(&sp[c4 * 4 + 1], hv.y);
        atomicAdd(&sp[c4 * 4 + 2], hv.z);
        atomicAdd(&sp[c4 * 4 + 3], hv.w);
    }
}

__global__ void pool_finish(const float* __restrict__ sums, const float* __restrict__ cnt,
                            __bf16* __restrict__ embbf) {
    int i = blockIdx.x * blockDim.x + threadIdx.x;
    if (i >= GG * 64) return;
    int g = i >> 6;
    embbf[i] = f2bf(sums[i] / fmaxf(cnt[g], 1.0f));
}

// Tiny MLP head: policy (G x 7) then value (G)
__global__ void head_kernel(const float* __restrict__ z, const float* __restrict__ polW,
                            const float* __restrict__ polb, const float* __restrict__ valW,
                            const float* __restrict__ valb, float* __restrict__ out) {
    int i = blockIdx.x * blockDim.x + threadIdx.x;
    if (i >= GG * 8) return;
    int g = i >> 3, j = i & 7;
    const float* zr = z + (size_t)g * HID;
    if (j < 7) {
        float acc = polb[j];
        #pragma unroll 8
        for (int k = 0; k < HID; ++k) acc = fmaf(zr[k], polW[k * 7 + j], acc);
        out[(size_t)g * 7 + j] = acc;
    } else {
        float acc = valb[0];
        #pragma unroll 8
        for (int k = 0; k < HID; ++k) acc = fmaf(zr[k], valW[k], acc);
        out[(size_t)GG * 7 + g] = tanhf(acc);
    }
}

// ---------------------------------------------------------------------------
extern "C" void kernel_launch(void* const* d_in, const int* in_sizes, int n_in,
                              void* d_out, int out_size, void* d_ws, size_t ws_size,
                              hipStream_t stream) {
    const float* x     = (const float*)d_in[0];
    const int*   eidx  = (const int*)  d_in[1];
    const int*   batch = (const int*)  d_in[2];
    const float* in_W  = (const float*)d_in[3];
    const float* in_b  = (const float*)d_in[4];
    const float* k_W   = (const float*)d_in[5];
    const float* k_b   = (const float*)d_in[6];
    const float* q_W   = (const float*)d_in[7];
    const float* q_b   = (const float*)d_in[8];
    const float* v_W   = (const float*)d_in[9];
    const float* v_b   = (const float*)d_in[10];
    const float* a_W   = (const float*)d_in[11];
    const float* a_b   = (const float*)d_in[12];
    const float* a_rel = (const float*)d_in[13];
    const float* m_rel = (const float*)d_in[14];
    const float* p_rel = (const float*)d_in[15];
    const float* skip  = (const float*)d_in[16];
    const float* fc1_W = (const float*)d_in[17];
    const float* fc1_b = (const float*)d_in[18];
    const float* pol_W = (const float*)d_in[19];
    const float* pol_b = (const float*)d_in[20];
    const float* val_W = (const float*)d_in[21];
    const float* val_b = (const float*)d_in[22];

    // ---- workspace carve (256B aligned) ----
    char* base = (char*)d_ws; size_t off = 0;
    auto carve = [&](size_t bytes) -> void* {
        void* p = base + off; off = (off + bytes + 255) & ~(size_t)255; return p;
    };
    float*  h      = (float*) carve((size_t)NN * 64 * 4);
    __bf16* hbf    = (__bf16*)carve((size_t)NN * 64 * 2);
    float*  q      = (float*) carve((size_t)NN * 64 * 4);
    __bf16* kbf    = (__bf16*)carve((size_t)NN * 64 * 2);
    __bf16* vbf    = (__bf16*)carve((size_t)NN * 64 * 2);
    float*  relbuf = (float*) carve((size_t)NN * 64 * 4);
    float*  logits = (float*) carve((size_t)RR * EE * 4 * 4);
    unsigned* maxb = (unsigned*)carve((size_t)NN * 4 * 4);
    float*  den    = (float*) carve((size_t)NN * 4 * 4);
    float*  agg    = (float*) carve((size_t)NN * 64 * 4);
    __bf16* gelubf = (__bf16*)carve((size_t)NN * 64 * 2);
    __bf16* wts    = (__bf16*)carve((size_t)(8 * 4096 + 32 * 4096 + 8192) * 2);
    float*  sums   = (float*) carve((size_t)GG * 64 * 4);
    float*  cnt    = (float*) carve((size_t)GG * 4);
    __bf16* embbf  = (__bf16*)carve((size_t)GG * 64 * 2);
    float*  z      = (float*) carve((size_t)GG * HID * 4);

    __bf16* kqvaT = wts;                 // [l][4 mats] each 4096
    __bf16* bdT   = wts + 8 * 4096;      // [(l*R+r)*2 + which] each 4096
    __bf16* fc1T  = wts + 40 * 4096;     // 128*64

    // ---- weight prep ----
    const float* Wsrc[8] = { k_W, q_W, v_W, a_W, k_W + 4096, q_W + 4096, v_W + 4096, a_W + 4096 };
    for (int m = 0; m < 8; ++m)
        transpose_w_bf16<<<(4096 + 255) / 256, 256, 0, stream>>>(Wsrc[m], kqvaT + m * 4096, 64);
    transpose_w_bf16<<<(8192 + 255) / 256, 256, 0, stream>>>(fc1_W, fc1T, HID);
    build_bd_bf16<<<(2 * RR * 2 * 4096 + 255) / 256, 256, 0, stream>>>(a_rel, m_rel, bdT);

    // ---- input projection ----
    input_proj<<<(NN + 255) / 256, 256, 0, stream>>>(x, in_W, in_b, h, hbf);

    dim3 gblk(32);
    const int nodeBlocks = (NN + 255) / 256;
    const int edgeBlocks = (EE + 255) / 256;

    for (int l = 0; l < 2; ++l) {
        __bf16* kT = kqvaT + (l * 4 + 0) * 4096;
        __bf16* qT = kqvaT + (l * 4 + 1) * 4096;
        __bf16* vT = kqvaT + (l * 4 + 2) * 4096;
        __bf16* aT = kqvaT + (l * 4 + 3) * 4096;

        // q (f32), k/v (bf16) projections via WMMA
        gemm64_wmma<4, 0, true, true, false><<<NN / 16, gblk, 0, stream>>>(
            hbf, qT, q_b + l * 64, q, nullptr, nullptr, nullptr);
        gemm64_wmma<4, 0, true, false, true><<<NN / 16, gblk, 0, stream>>>(
            hbf, kT, k_b + l * 64, nullptr, kbf, nullptr, nullptr);
        gemm64_wmma<4, 0, true, false, true><<<NN / 16, gblk, 0, stream>>>(
            hbf, vT, v_b + l * 64, nullptr, vbf, nullptr, nullptr);

        hipMemsetAsync(maxb, 0, (size_t)NN * 4 * 4, stream);
        hipMemsetAsync(den,  0, (size_t)NN * 4 * 4, stream);
        hipMemsetAsync(agg,  0, (size_t)NN * 64 * 4, stream);

        // pass 1: per relation, k_rel GEMM (block-diag) then logits + segment max
        for (int r = 0; r < RR; ++r) {
            __bf16* bdA = bdT + ((l * RR + r) * 2 + 0) * 4096;
            gemm64_wmma<4, 0, false, true, false><<<NN / 16, gblk, 0, stream>>>(
                kbf, bdA, nullptr, relbuf, nullptr, nullptr, nullptr);
            edge_logits<<<edgeBlocks, 256, 0, stream>>>(eidx + r * 2 * EE, q, relbuf,
                                                        p_rel + (l * RR + r) * 4,
                                                        logits + (size_t)r * EE * 4, maxb);
        }
        // pass 2: softmax denominators over all R*E edges
        edge_softmax_den<<<(RR * EE + 255) / 256, 256, 0, stream>>>(eidx, maxb, logits, den);
        // pass 3: per relation, v_rel GEMM then weighted scatter
        for (int r = 0; r < RR; ++r) {
            __bf16* bdM = bdT + ((l * RR + r) * 2 + 1) * 4096;
            gemm64_wmma<4, 0, false, true, false><<<NN / 16, gblk, 0, stream>>>(
                vbf, bdM, nullptr, relbuf, nullptr, nullptr, nullptr);
            edge_agg<<<edgeBlocks, 256, 0, stream>>>(eidx + r * 2 * EE,
                                                     logits + (size_t)r * EE * 4, den, relbuf, agg);
        }
        // out = gelu(agg) @ a_W + a_b ; h = g*out + (1-g)*h   (g = sigmoid(skip[l]))
        gelu_to_bf16<<<(NN * 64 + 255) / 256, 256, 0, stream>>>(agg, gelubf, NN * 64);
        gemm64_wmma<4, 2, true, true, true><<<NN / 16, gblk, 0, stream>>>(
            gelubf, aT, a_b + l * 64, h, hbf, h, skip + l);
    }

    // ---- segment-mean pool + head ----
    hipMemsetAsync(sums, 0, (size_t)GG * 64 * 4, stream);
    hipMemsetAsync(cnt,  0, (size_t)GG * 4, stream);
    pool_accum<<<nodeBlocks, 256, 0, stream>>>(h, batch, sums, cnt);
    pool_finish<<<(GG * 64 + 255) / 256, 256, 0, stream>>>(sums, cnt, embbf);
    gemm64_wmma<8, 1, true, true, false><<<GG / 16, gblk, 0, stream>>>(
        embbf, fc1T, fc1_b, z, nullptr, nullptr, nullptr);
    head_kernel<<<(GG * 8 + 255) / 256, 256, 0, stream>>>(z, pol_W, pol_b, val_W, val_b, (float*)d_out);
}
